// GMS_32401233281698
// MI455X (gfx1250) — matrix operand order repaired and checked by reading, not compiled, not run on hardware
//
#include <hip/hip_runtime.h>
#include <hip/hip_bf16.h>

#define DIMV     128
#define NVARS    50000
#define NLITS    100000
#define NCLAUSES 200000
#define NCELLS   600000
#define NROUNDS  8

// gfx1250 async Global->LDS staging (inline asm; flip to 0 to fall back to
// plain ds-store staging if the assembler rejects the mnemonic)
#define USE_ASYNC_LDS 1

typedef __attribute__((ext_vector_type(16))) _Float16 v16h;
typedef __attribute__((ext_vector_type(8)))  _Float16 v8h;
typedef __attribute__((ext_vector_type(8)))  float    v8f;

// ---------------------------------------------------------------------------
// Async copy helpers: one b128 (16B) per lane per issue, ASYNCcnt-tracked.
// ISA 10.2: flat LDS address low 32 bits == wave-relative LDS byte offset,
// and HW adds LDS_BASE to the VDST VGPR for async-to-LDS ops.
// ---------------------------------------------------------------------------
__device__ __forceinline__ void async_ld_b128(unsigned int lds_addr, const void* gptr) {
#if USE_ASYNC_LDS
  asm volatile("global_load_async_to_lds_b128 %0, %1, off"
               :: "v"(lds_addr), "v"((unsigned long long)(uintptr_t)gptr)
               : "memory");
#endif
}
__device__ __forceinline__ void async_wait0() {
#if USE_ASYNC_LDS
  asm volatile("s_wait_asynccnt 0x0" ::: "memory");
#endif
}

// ---------------------------------------------------------------------------
// WMMA fragment helpers (CDNA5 v_wmma_f32_16x16x32_f16 layouts, wave32)
// A 16x32 f16: lane m=lane&15; halfs 0..7 = K kb*32+kbase..+7, halfs 8..15 =
//              K kb*32+kbase+16..+23, kbase = (lane<16)?0:8.
// B 32x16 f16: lane n=lane&15; halfs 0..7 = K kb*32+koff..+7, halfs 8..15 =
//              K kb*32+koff+16..+23, koff = (lane<16)?0:8  ->  B[k][n] = W[n][k] (row-major W!).
// C/D 16x16 f32: VGPR r holds (M = r + ((lane<16)?0:8), N = lane&15).
// ---------------------------------------------------------------------------
__device__ __forceinline__ v16h load_a_frag(const _Float16* __restrict__ A, int lda,
                                            int kb, int lane) {
  const int m  = lane & 15;
  const int kA = kb * 32 + ((lane < 16) ? 0 : 8);
  const _Float16* p = A + m * lda + kA;
  v8h lo = *(const v8h*)(p);
  v8h hi = *(const v8h*)(p + 16);
  v16h a;
#pragma unroll
  for (int i = 0; i < 8; ++i) { a[i] = lo[i]; a[i + 8] = hi[i]; }
  return a;
}

__device__ __forceinline__ v16h load_b_frag(const _Float16* __restrict__ W, int ldw,
                                            int ncol, int kb, int lane) {
  const int nn = lane & 15;
  const int kB = kb * 32 + ((lane < 16) ? 0 : 8);
  const _Float16* q = W + (size_t)(ncol + nn) * ldw + kB;
  v8h lo = *(const v8h*)(q);
  v8h hi = *(const v8h*)(q + 16);
  v16h b;
#pragma unroll
  for (int i = 0; i < 8; ++i) { b[i] = lo[i]; b[i + 8] = hi[i]; }
  return b;
}

__device__ __forceinline__ v8f mm_tile_1(const _Float16* __restrict__ A, int lda, int nkb,
                                         const _Float16* __restrict__ W, int ldw, int ncol,
                                         v8f acc, int lane) {
  for (int kb = 0; kb < nkb; ++kb) {
    v16h b = load_b_frag(W, ldw, ncol, kb, lane);
    v16h a = load_a_frag(A, lda, kb, lane);
    acc = __builtin_amdgcn_wmma_f32_16x16x32_f16(false, a, false, b, (short)0, acc,
                                                 false, false);
  }
  return acc;
}

__device__ __forceinline__ void mm_tile_2(const _Float16* __restrict__ A, int lda, int nkb,
                                          const _Float16* __restrict__ W, int ldw, int ncol,
                                          float bv, int lane, v8f& acc0, v8f& acc1) {
#pragma unroll
  for (int r = 0; r < 8; ++r) { acc0[r] = bv; acc1[r] = bv; }
  for (int kb = 0; kb < nkb; ++kb) {
    v16h b  = load_b_frag(W, ldw, ncol, kb, lane);
    v16h a0 = load_a_frag(A, lda, kb, lane);
    v16h a1 = load_a_frag(A + 16 * lda, lda, kb, lane);
    acc0 = __builtin_amdgcn_wmma_f32_16x16x32_f16(false, a0, false, b, (short)0, acc0,
                                                  false, false);
    acc1 = __builtin_amdgcn_wmma_f32_16x16x32_f16(false, a1, false, b, (short)0, acc1,
                                                  false, false);
  }
}

// ---------------------------------------------------------------------------
// Fused 3-layer MLP (128->128 relu ->128 relu ->128), 32 rows per block.
// Input is the f16 shadow state; output f32 (feeds the atomic scatter).
// ---------------------------------------------------------------------------
__global__ void __launch_bounds__(256)
mlp3_kernel(const _Float16* __restrict__ X16,
            const _Float16* __restrict__ W1, const float* __restrict__ B1,
            const _Float16* __restrict__ W2, const float* __restrict__ B2,
            const _Float16* __restrict__ W3, const float* __restrict__ B3,
            float* __restrict__ Y) {
  __shared__ _Float16 A0[32][136];   // row stride 272B (16B-aligned)
  __shared__ _Float16 A1[32][136];
  const int tid  = threadIdx.x;
  const int lane = tid & 31;
  const int wave = tid >> 5;
  const int n    = lane & 15;
  const int mb   = (lane < 16) ? 0 : 8;
  const size_t row0 = (size_t)blockIdx.x * 32;

#if USE_ASYNC_LDS
  {  // 32 rows x 256B, 16B chunks: 512 chunks, 2 per thread
    const unsigned int lds0 = (unsigned int)(uintptr_t)&A0[0][0];
#pragma unroll
    for (int c = tid; c < 512; c += 256) {
      int row = c >> 4, seg = c & 15;
      async_ld_b128(lds0 + row * 272 + seg * 16,
                    X16 + (row0 + row) * DIMV + seg * 8);
    }
    async_wait0();
  }
#else
  for (int i = tid; i < 32 * 128; i += 256) {
    int m = i >> 7, k = i & 127;
    A0[m][k] = X16[(row0 + m) * DIMV + k];
  }
#endif
  __syncthreads();

  v8f acc0, acc1;
  // layer 1: A0 -> A1 (ReLU)
  mm_tile_2(&A0[0][0], 136, 4, W1, 128, wave * 16, B1[wave * 16 + n], lane, acc0, acc1);
#pragma unroll
  for (int r = 0; r < 8; ++r) {
    float v0 = acc0[r] > 0.f ? acc0[r] : 0.f;
    float v1 = acc1[r] > 0.f ? acc1[r] : 0.f;
    A1[mb + r][wave * 16 + n]      = (_Float16)v0;
    A1[16 + mb + r][wave * 16 + n] = (_Float16)v1;
  }
  __syncthreads();
  // layer 2: A1 -> A0 (ReLU)
  mm_tile_2(&A1[0][0], 136, 4, W2, 128, wave * 16, B2[wave * 16 + n], lane, acc0, acc1);
#pragma unroll
  for (int r = 0; r < 8; ++r) {
    float v0 = acc0[r] > 0.f ? acc0[r] : 0.f;
    float v1 = acc1[r] > 0.f ? acc1[r] : 0.f;
    A0[mb + r][wave * 16 + n]      = (_Float16)v0;
    A0[16 + mb + r][wave * 16 + n] = (_Float16)v1;
  }
  __syncthreads();
  // layer 3: A0 -> Y (no ReLU), f32 out
  mm_tile_2(&A0[0][0], 136, 4, W3, 128, wave * 16, B3[wave * 16 + n], lane, acc0, acc1);
#pragma unroll
  for (int r = 0; r < 8; ++r) {
    Y[(row0 + mb + r) * DIMV + wave * 16 + n]      = acc0[r];
    Y[(row0 + 16 + mb + r) * DIMV + wave * 16 + n] = acc1[r];
  }
}

// ---------------------------------------------------------------------------
// Fused LSTM cell: gates = x@Wih^T + h@Whh^T + bih + bhh (WMMA, f32 acc in
// LDS), then fused sigmoid/tanh update. h kept as f16 shadow; c stays f32.
// KX=128: clause side (x = LC, f32). KX=256: literal side (x cols 0..127 = CL
// f32; cols 128..255 = flip rows of the f16 L_h shadow, async-copied).
// ---------------------------------------------------------------------------
template <int KX>
__global__ void __launch_bounds__(256)
lstm_kernel(const float* __restrict__ X1, const _Float16* __restrict__ X2_16,
            const _Float16* __restrict__ Hin16, _Float16* __restrict__ Hout16,
            float* __restrict__ Cst,
            const _Float16* __restrict__ Wih, const _Float16* __restrict__ Whh,
            const float* __restrict__ Bih, const float* __restrict__ Bhh,
            int n_half) {
  __shared__ _Float16 Ax[16][KX + 8];   // stride (KX+8)*2 B: 272 or 528, 16B-aligned
  __shared__ _Float16 Ah[16][136];
  __shared__ float    gates[16][512];
  const int tid  = threadIdx.x;
  const int lane = tid & 31;
  const int wave = tid >> 5;
  const int n    = lane & 15;
  const int mb   = (lane < 16) ? 0 : 8;
  const size_t row0 = (size_t)blockIdx.x * 16;

  // x low half: f32 -> f16 via VGPRs
  for (int i = tid; i < 16 * 128; i += 256) {
    int m = i >> 7, k = i & 127;
    Ax[m][k] = (_Float16)X1[(row0 + m) * DIMV + k];
  }
#if USE_ASYNC_LDS
  if (KX == 256) {  // flip half: raw f16 copy, 16 rows x 16 chunks
    int m = tid >> 4, seg = tid & 15;
    size_t rr = row0 + (size_t)m;
    size_t fr = (rr < (size_t)n_half) ? rr + n_half : rr - n_half;
    async_ld_b128((unsigned int)(uintptr_t)&Ax[0][0] + m * ((KX + 8) * 2) + 256 + seg * 16,
                  X2_16 + fr * DIMV + seg * 8);
  }
  {  // h: raw f16 copy, 16 rows x 16 chunks
    int m = tid >> 4, seg = tid & 15;
    async_ld_b128((unsigned int)(uintptr_t)&Ah[0][0] + m * 272 + seg * 16,
                  Hin16 + (row0 + m) * DIMV + seg * 8);
  }
  async_wait0();
#else
  if (KX == 256) {
    for (int i = tid; i < 16 * 128; i += 256) {
      int m = i >> 7, k = i & 127;
      size_t rr = row0 + (size_t)m;
      size_t fr = (rr < (size_t)n_half) ? rr + n_half : rr - n_half;
      Ax[m][128 + k] = X2_16[fr * DIMV + k];
    }
  }
  for (int i = tid; i < 16 * 128; i += 256) {
    int m = i >> 7, k = i & 127;
    Ah[m][k] = Hin16[(row0 + m) * DIMV + k];
  }
#endif
  __syncthreads();

#pragma unroll
  for (int tt = 0; tt < 4; ++tt) {
    const int ncol = (wave * 4 + tt) * 16;
    float bv = Bih[ncol + n] + Bhh[ncol + n];
    v8f acc;
#pragma unroll
    for (int r = 0; r < 8; ++r) acc[r] = bv;
    acc = mm_tile_1(&Ax[0][0], KX + 8, KX / 32, Wih, KX, ncol, acc, lane);
    acc = mm_tile_1(&Ah[0][0], 136, 4, Whh, 128, ncol, acc, lane);
#pragma unroll
    for (int r = 0; r < 8; ++r) gates[mb + r][ncol + n] = acc[r];
  }
  __syncthreads();

  for (int i = tid; i < 16 * 128; i += 256) {
    int m = i >> 7, d = i & 127;
    float ig = gates[m][d], fg = gates[m][128 + d];
    float gg = gates[m][256 + d], og = gates[m][384 + d];
    size_t gi = (row0 + m) * DIMV + d;
    float cold = Cst[gi];
    float si = 1.f / (1.f + __expf(-ig));
    float sf = 1.f / (1.f + __expf(-fg));
    float so = 1.f / (1.f + __expf(-og));
    float cn = sf * cold + si * tanhf(gg);
    Cst[gi]   = cn;
    Hout16[gi] = (_Float16)(so * tanhf(cn));
  }
}

// ---------------------------------------------------------------------------
// Final vote MLP: x=[Lh[v], Lh[v+n_vars]] (256) ->128 relu ->128 relu ->1
// ---------------------------------------------------------------------------
__global__ void __launch_bounds__(256)
vote_kernel(const _Float16* __restrict__ Lh16,
            const _Float16* __restrict__ W1, const float* __restrict__ B1,
            const _Float16* __restrict__ W2, const float* __restrict__ B2,
            const float* __restrict__ W3, const float* __restrict__ B3,
            float* __restrict__ out, int n_vars) {
  __shared__ _Float16 A0[16][264];   // stride 528B, 16B-aligned
  __shared__ _Float16 A1[16][136];
  __shared__ float    red[16][16];
  const int tid  = threadIdx.x;
  const int lane = tid & 31;
  const int wave = tid >> 5;
  const int n    = lane & 15;
  const int mb   = (lane < 16) ? 0 : 8;
  const size_t row0 = (size_t)blockIdx.x * 16;

#if USE_ASYNC_LDS
  {  // 16 rows x 512B (two 256B halves), 512 chunks, 2 per thread
    const unsigned int lds0 = (unsigned int)(uintptr_t)&A0[0][0];
#pragma unroll
    for (int c = tid; c < 512; c += 256) {
      int m = c >> 5, seg = c & 31;
      const _Float16* g = (seg < 16)
          ? Lh16 + (row0 + m) * DIMV + seg * 8
          : Lh16 + (row0 + m + (size_t)n_vars) * DIMV + (seg - 16) * 8;
      async_ld_b128(lds0 + m * 528 + seg * 16, g);
    }
    async_wait0();
  }
#else
  for (int i = tid; i < 16 * 256; i += 256) {
    int m = i >> 8, k = i & 255;
    A0[m][k] = (k < 128) ? Lh16[(row0 + m) * DIMV + k]
                         : Lh16[(row0 + m + (size_t)n_vars) * DIMV + (k - 128)];
  }
#endif
  __syncthreads();

  v8f acc;
  float bv = B1[wave * 16 + n];
#pragma unroll
  for (int r = 0; r < 8; ++r) acc[r] = bv;
  acc = mm_tile_1(&A0[0][0], 264, 8, W1, 256, wave * 16, acc, lane);
#pragma unroll
  for (int r = 0; r < 8; ++r) {
    float v = acc[r] > 0.f ? acc[r] : 0.f;
    A1[mb + r][wave * 16 + n] = (_Float16)v;
  }
  __syncthreads();

  bv = B2[wave * 16 + n];
#pragma unroll
  for (int r = 0; r < 8; ++r) acc[r] = bv;
  acc = mm_tile_1(&A1[0][0], 136, 4, W2, 128, wave * 16, acc, lane);
#pragma unroll
  for (int r = 0; r < 8; ++r) {
    float v = acc[r] > 0.f ? acc[r] : 0.f;
    A0[mb + r][wave * 16 + n] = (_Float16)v;
  }
  __syncthreads();

  {  // layer 3: [16x128] . w3[128] + b3
    int m = tid >> 4, s = tid & 15;
    float p = 0.f;
#pragma unroll
    for (int j = 0; j < 8; ++j) {
      int d = s * 8 + j;
      p += (float)A0[m][d] * W3[d];
    }
    red[m][s] = p;
  }
  __syncthreads();
  if (tid < 16) {
    float t = B3[0];
#pragma unroll
    for (int s = 0; s < 16; ++s) t += red[tid][s];
    out[row0 + tid] = t;
  }
}

// ---------------------------------------------------------------------------
// Utility kernels
// ---------------------------------------------------------------------------
__global__ void zero_kernel(float* __restrict__ p, size_t ntot) {
  size_t i = (size_t)blockIdx.x * blockDim.x + threadIdx.x;
  size_t st = (size_t)gridDim.x * blockDim.x;
  for (; i < ntot; i += st) p[i] = 0.f;
}

__global__ void cvt_f16_kernel(const float* __restrict__ in, _Float16* __restrict__ out,
                               int ntot) {
  int i = blockIdx.x * 256 + threadIdx.x;
  if (i < ntot) out[i] = (_Float16)in[i];
}

__global__ void init_state_kernel(const float* __restrict__ w, const float* __restrict__ b,
                                  _Float16* __restrict__ h16, float* __restrict__ c,
                                  size_t ntot) {
  size_t i = (size_t)blockIdx.x * blockDim.x + threadIdx.x;
  size_t st = (size_t)gridDim.x * blockDim.x;
  for (; i < ntot; i += st) {
    int d = (int)(i & 127);
    h16[i] = (_Float16)(w[d] + b[d]);  // init scalar == 1.0
    c[i] = 0.f;
  }
}

// segment scatter-add: 32 threads per cell, 4 floats per thread
__global__ void __launch_bounds__(256)
scatter_add_kernel(const float* __restrict__ src, const int* __restrict__ gidx,
                   const int* __restrict__ sidx, float* __restrict__ dst, int n_cells) {
  size_t t = (size_t)blockIdx.x * 256 + threadIdx.x;
  size_t cell = t >> 5;
  if (cell >= (size_t)n_cells) return;
  int q = (int)(t & 31);
  int g = gidx[cell], s = sidx[cell];
  const float4 v = *((const float4*)(src + (size_t)g * DIMV) + q);
  float* dp = dst + (size_t)s * DIMV + q * 4;
  atomicAdd(dp + 0, v.x);
  atomicAdd(dp + 1, v.y);
  atomicAdd(dp + 2, v.z);
  atomicAdd(dp + 3, v.w);
}

// ---------------------------------------------------------------------------
extern "C" void kernel_launch(void* const* d_in, const int* in_sizes, int n_in,
                              void* d_out, int out_size, void* d_ws, size_t ws_size,
                              hipStream_t stream) {
  (void)in_sizes; (void)n_in; (void)out_size; (void)ws_size;
  // input order: 0 Li_w 1 Li_b 2 Ci_w 3 Ci_b | 4..9 Lm w1,b1,w2,b2,w3,b3 |
  // 10..15 Cm | 16..21 V w1,b1,w2,b2,w3,b3 | 22 Cu_wih 23 Cu_bih 24 Cu_whh
  // 25 Cu_bhh | 26 Lu_wih 27 Lu_bih 28 Lu_whh 29 Lu_bhh | 30 lit 31 clause 32 n_vars
  #define FP(i) ((const float*)d_in[(i)])
  const int* lit_idx    = (const int*)d_in[30];
  const int* clause_idx = (const int*)d_in[31];

  char* base = (char*)d_ws;
  size_t off = 0;
  auto carve = [&](size_t bytes) -> void* {
    void* p = base + off;
    off += (bytes + 255) & ~(size_t)255;
    return p;
  };
  _Float16* Lh16A = (_Float16*)carve((size_t)NLITS * DIMV * 2);     // f16 h shadows
  _Float16* Lh16B = (_Float16*)carve((size_t)NLITS * DIMV * 2);
  _Float16* Ch16  = (_Float16*)carve((size_t)NCLAUSES * DIMV * 2);
  float* Lc   = (float*)carve((size_t)NLITS * DIMV * 4);
  float* Cc   = (float*)carve((size_t)NCLAUSES * DIMV * 4);
  float* BufL = (float*)carve((size_t)NLITS * DIMV * 4);     // L_pre / CL (f32)
  float* BufC = (float*)carve((size_t)NCLAUSES * DIMV * 4);  // LC / C_pre (f32)
  _Float16* wLm1  = (_Float16*)carve(128 * 128 * 2);
  _Float16* wLm2  = (_Float16*)carve(128 * 128 * 2);
  _Float16* wLm3  = (_Float16*)carve(128 * 128 * 2);
  _Float16* wCm1  = (_Float16*)carve(128 * 128 * 2);
  _Float16* wCm2  = (_Float16*)carve(128 * 128 * 2);
  _Float16* wCm3  = (_Float16*)carve(128 * 128 * 2);
  _Float16* wCuIH = (_Float16*)carve(512 * 128 * 2);
  _Float16* wCuHH = (_Float16*)carve(512 * 128 * 2);
  _Float16* wLuIH = (_Float16*)carve(512 * 256 * 2);
  _Float16* wLuHH = (_Float16*)carve(512 * 128 * 2);
  _Float16* wV1   = (_Float16*)carve(128 * 256 * 2);
  _Float16* wV2   = (_Float16*)carve(128 * 128 * 2);

  auto cvt = [&](int idx, _Float16* dst, int nel) {
    cvt_f16_kernel<<<(nel + 255) / 256, 256, 0, stream>>>(FP(idx), dst, nel);
  };
  cvt(4, wLm1, 128 * 128);  cvt(6, wLm2, 128 * 128);  cvt(8, wLm3, 128 * 128);
  cvt(10, wCm1, 128 * 128); cvt(12, wCm2, 128 * 128); cvt(14, wCm3, 128 * 128);
  cvt(22, wCuIH, 512 * 128); cvt(24, wCuHH, 512 * 128);
  cvt(26, wLuIH, 512 * 256); cvt(28, wLuHH, 512 * 128);
  cvt(16, wV1, 128 * 256);   cvt(18, wV2, 128 * 128);

  init_state_kernel<<<4096, 256, 0, stream>>>(FP(0), FP(1), Lh16A, Lc,
                                              (size_t)NLITS * DIMV);
  init_state_kernel<<<4096, 256, 0, stream>>>(FP(2), FP(3), Ch16, Cc,
                                              (size_t)NCLAUSES * DIMV);

  const int scatter_blocks = (int)(((size_t)NCELLS * 32 + 255) / 256);
  for (int r = 0; r < NROUNDS; ++r) {
    const _Float16* Lh16R = (r & 1) ? Lh16B : Lh16A;
    _Float16* Lh16W       = (r & 1) ? Lh16A : Lh16B;

    // L_pre = MLP(L_h)
    mlp3_kernel<<<NLITS / 32, 256, 0, stream>>>(Lh16R, wLm1, FP(5), wLm2, FP(7),
                                                wLm3, FP(9), BufL);
    // LC = segment_sum(L_pre[lit_idx] -> clause_idx)
    zero_kernel<<<2048, 256, 0, stream>>>(BufC, (size_t)NCLAUSES * DIMV);
    scatter_add_kernel<<<scatter_blocks, 256, 0, stream>>>(BufL, lit_idx, clause_idx,
                                                           BufC, NCELLS);
    // C_h, C_c = LSTM(LC, C_h, C_c)
    lstm_kernel<128><<<NCLAUSES / 16, 256, 0, stream>>>(BufC, (const _Float16*)nullptr,
                                                        Ch16, Ch16, Cc,
                                                        wCuIH, wCuHH, FP(23), FP(25), 0);
    // C_pre = MLP(C_h)
    mlp3_kernel<<<NCLAUSES / 32, 256, 0, stream>>>(Ch16, wCm1, FP(11), wCm2, FP(13),
                                                   wCm3, FP(15), BufC);
    // CL = segment_sum(C_pre[clause_idx] -> lit_idx)
    zero_kernel<<<2048, 256, 0, stream>>>(BufL, (size_t)NLITS * DIMV);
    scatter_add_kernel<<<scatter_blocks, 256, 0, stream>>>(BufC, clause_idx, lit_idx,
                                                           BufL, NCELLS);
    // L_h, L_c = LSTM([CL, flip(L_h)], L_h, L_c)   (double-buffered L_h shadow)
    lstm_kernel<256><<<NLITS / 16, 256, 0, stream>>>(BufL, Lh16R, Lh16R, Lh16W, Lc,
                                                     wLuIH, wLuHH, FP(27), FP(29),
                                                     NVARS);
  }
  // after 8 rounds (last write on odd r) final L_h is in Lh16A
  vote_kernel<<<NVARS / 16, 256, 0, stream>>>(Lh16A, wV1, FP(17), wV2, FP(19),
                                              FP(20), FP(21), (float*)d_out, NVARS);
  #undef FP
}
